// Attention_30880814859134
// MI455X (gfx1250) — matrix operand order repaired and checked by reading, not compiled, not run on hardware
//
#include <hip/hip_runtime.h>
#include <hip/hip_bf16.h>

// ---------------- problem constants ----------------
#define BB 8
#define NN 1024
#define CC 768
#define HH 12
#define DD 64
#define SCALE 0.125f      // DD^-0.5
#define EPS 1e-6f

typedef unsigned short u16;
typedef u16  u16x8  __attribute__((ext_vector_type(8)));
typedef u16  u16x16 __attribute__((ext_vector_type(16)));
typedef __bf16 v16bf __attribute__((ext_vector_type(16)));
typedef float  v8f   __attribute__((ext_vector_type(8)));

// fp32 -> bf16 round-to-nearest-even
static __device__ __forceinline__ u16 f2bf(float f) {
  unsigned u = __float_as_uint(f);
  unsigned r = u + 0x7FFFu + ((u >> 16) & 1u);
  return (u16)(r >> 16);
}

// A-fragment (16x32 bf16, M x K): lane (g=l>>4, m=l&15) holds
// elems[0..15] = A[m][16g .. 16g+16)
// caller passes p = &A[m][16g]; reads p[0..7] and p[8..15].
static __device__ __forceinline__ v16bf load_a_frag(const u16* p) {
  u16x8 lo = *(const u16x8*)(p);
  u16x8 hi = *(const u16x8*)(p + 8);
  u16x16 w;
#pragma unroll
  for (int i = 0; i < 8; ++i) { w[i] = lo[i]; w[i + 8] = hi[i]; }
  return __builtin_bit_cast(v16bf, w);
}

// B-fragment (32x16 bf16, K x N): lane (g, n=l&15) holds B[16g+j][n], j=0..15
// caller passes p = &BT[n][16g]  (32 bytes contiguous, 32B aligned)
static __device__ __forceinline__ v16bf load_b_frag(const u16* p) {
  u16x16 w = *(const u16x16*)p;
  return __builtin_bit_cast(v16bf, w);
}

static __device__ __forceinline__ v8f wmma_bf16(v16bf a, v16bf b, v8f c) {
  return __builtin_amdgcn_wmma_f32_16x16x32_bf16(false, a, false, b,
                                                 (short)0, c, false, false);
}

// reductions across the 16-lane half (bits 0..3 of lane id)
static __device__ __forceinline__ float rmax16(float x) {
#pragma unroll
  for (int m = 1; m < 16; m <<= 1) x = fmaxf(x, __shfl_xor(x, m, 32));
  return x;
}
static __device__ __forceinline__ float rsum16(float x) {
#pragma unroll
  for (int m = 1; m < 16; m <<= 1) x += __shfl_xor(x, m, 32);
  return x;
}

// CDNA5 async global -> LDS copy, 16 bytes per lane (VGLOBAL "GV" mode).
static __device__ __forceinline__ void async_ld_b128(unsigned lds_off,
                                                     const u16* g) {
  unsigned long long ga = (unsigned long long)(uintptr_t)g;
  asm volatile("global_load_async_to_lds_b128 %0, %1, off"
               :: "v"(lds_off), "v"(ga) : "memory");
}
static __device__ __forceinline__ void wait_async0() {
  asm volatile("s_wait_asynccnt 0" ::: "memory");
}
static __device__ __forceinline__ unsigned lds_off_of(const void* p) {
  return (unsigned)(uintptr_t)p;    // LDS aperture: addr[31:0] == LDS offset
}

// ---------------- kernel 0: fp32 -> bf16 cast ----------------
__global__ void cast_bf16_kernel(const float* __restrict__ src,
                                 u16* __restrict__ dst, int n) {
  int i = blockIdx.x * 256 + threadIdx.x;
  if (i < n) dst[i] = f2bf(src[i]);
}

// ======================================================================
// Block-tiled GEMM core: 4 waves, block tile M=64 x N=64, K stepped by 32.
// Double-buffered LDS staging of A (64x32) and B (64x32, BT rows) via
// async b128 loads; tile k+1 is in flight while tile k feeds the WMMAs.
// Each wave computes a 16x64 slice (4 WMMAs / k-step; 16 per block).
// acc[t][r] -> row m0 + w*16 + 8*half + r, col n0 + t*16 + ln.
// ======================================================================
#define GEMM_BODY(ABASE, BBASE)                                              \
  __shared__ __attribute__((aligned(32))) u16 lA[2][64 * 32];                \
  __shared__ __attribute__((aligned(32))) u16 lB[2][64 * 32];                \
  const int tid = threadIdx.x;                                               \
  const int w = tid >> 5, l = tid & 31, half = (l >> 4), ln = l & 15;        \
  const int m0 = blockIdx.x * 64;                                            \
  const int n0 = blockIdx.y * 64;                                            \
  /* staging assignment: row = tid>>1 (0..63), 32B half = tid&1 */           \
  const int srow = tid >> 1, shalf = tid & 1;                                \
  const u16* gA = (ABASE) + (size_t)(m0 + srow) * CC + shalf * 16;           \
  const u16* gB = (BBASE) + (size_t)(n0 + srow) * CC + shalf * 16;           \
  const unsigned loA0 = lds_off_of(&lA[0][srow * 32 + shalf * 16]);          \
  const unsigned loB0 = lds_off_of(&lB[0][srow * 32 + shalf * 16]);          \
  const unsigned bufb = 64 * 32 * 2; /* bytes per LDS buffer */              \
  v8f acc[4] = {};                                                           \
  /* prologue: stage k-tile 0 into buffer 0 */                               \
  async_ld_b128(loA0, gA);                                                   \
  async_ld_b128(loA0 + 16, gA + 8);                                          \
  async_ld_b128(loB0, gB);                                                   \
  async_ld_b128(loB0 + 16, gB + 8);                                          \
  _Pragma("unroll 1")                                                        \
  for (int it = 0; it < CC / 32; ++it) {                                     \
    wait_async0();            /* only tile `it` is outstanding */            \
    __syncthreads();          /* tile `it` visible to all waves  */          \
    const int cur = it & 1;                                                  \
    if (it + 1 < CC / 32) {   /* prefetch tile it+1 into other buffer */     \
      const unsigned nb = (unsigned)((it + 1) & 1) * bufb;                   \
      const int k1 = (it + 1) * 32;                                          \
      async_ld_b128(loA0 + nb, gA + k1);                                     \
      async_ld_b128(loA0 + nb + 16, gA + k1 + 8);                            \
      async_ld_b128(loB0 + nb, gB + k1);                                     \
      async_ld_b128(loB0 + nb + 16, gB + k1 + 8);                            \
    }                                                                        \
    v16bf a  = load_a_frag(&lA[cur][(w * 16 + ln) * 32 + 16 * half]);        \
    v16bf b0 = load_b_frag(&lB[cur][(0 * 16 + ln) * 32 + 16 * half]);        \
    v16bf b1 = load_b_frag(&lB[cur][(1 * 16 + ln) * 32 + 16 * half]);        \
    v16bf b2 = load_b_frag(&lB[cur][(2 * 16 + ln) * 32 + 16 * half]);        \
    v16bf b3 = load_b_frag(&lB[cur][(3 * 16 + ln) * 32 + 16 * half]);        \
    acc[0] = wmma_bf16(a, b0, acc[0]);                                       \
    acc[1] = wmma_bf16(a, b1, acc[1]);                                       \
    acc[2] = wmma_bf16(a, b2, acc[2]);                                       \
    acc[3] = wmma_bf16(a, b3, acc[3]);                                       \
  }

// ---------------- kernel 1: QKV GEMM ----------------
// qkv[m][o] = sum_c x[m][c] * w_qkv[o][c];  grid (128, 36), block 128.
// tileN 0..11 -> Q (scaled, [B,H,N,D]), 12..23 -> K, 24..35 -> V^T [B,H,D,N].
__global__ __launch_bounds__(128)
void gemm_qkv_kernel(const u16* __restrict__ xb, const u16* __restrict__ wb,
                     u16* __restrict__ qbuf, u16* __restrict__ kbuf,
                     u16* __restrict__ vtbuf) {
  GEMM_BODY(xb, wb)
  const int sec = n0 / CC;             // uniform per block: 0=Q 1=K 2=V
#pragma unroll
  for (int t = 0; t < 4; ++t) {
    int o = n0 + t * 16 + ln;
    int c = o - sec * CC;
    int h = c >> 6, d = c & 63;
#pragma unroll
    for (int r = 0; r < 8; ++r) {
      int m = m0 + w * 16 + 8 * half + r;
      int b = m >> 10, tok = m & 1023;
      float v = acc[t][r];
      if (sec == 0)
        qbuf[(((size_t)(b * HH + h) << 10) + tok) * DD + d] = f2bf(v * SCALE);
      else if (sec == 1)
        kbuf[(((size_t)(b * HH + h) << 10) + tok) * DD + d] = f2bf(v);
      else
        vtbuf[(((size_t)(b * HH + h) * DD + d) << 10) + tok] = f2bf(v);
    }
  }
}

// ---------------- kernel 3: output projection ----------------
// out[m][o] = sum_c ao[m][c] * w_proj[o][c] + b_proj[o]; grid (128,12).
__global__ __launch_bounds__(128)
void gemm_proj_kernel(const u16* __restrict__ ao, const u16* __restrict__ wpb,
                      const float* __restrict__ bias, float* __restrict__ out) {
  GEMM_BODY(ao, wpb)
#pragma unroll
  for (int t = 0; t < 4; ++t) {
    int o = n0 + t * 16 + ln;
    float bi = bias[o];
#pragma unroll
    for (int r = 0; r < 8; ++r) {
      int m = m0 + w * 16 + 8 * half + r;
      out[(size_t)m * CC + o] = acc[t][r] + bi;
    }
  }
}

// ---------------- kernel 2: masked flash attention ----------------
// grid (64, 96), block 32. One wave: 16 query rows of one (b,h), full D=64.
__global__ __launch_bounds__(32)
void attn_kernel(const u16* __restrict__ q, const u16* __restrict__ k,
                 const u16* __restrict__ vt, const float* __restrict__ policy,
                 u16* __restrict__ ao) {
  __shared__ __attribute__((aligned(32))) u16 pt[16 * 32];  // P tile bf16

  const int l = threadIdx.x, half = l >> 4, ln = l & 15;
  const int bh = blockIdx.y;            // b*12 + h
  const int b = bh / HH, h = bh - b * HH;
  const int qm0 = blockIdx.x * 16;
  const u16* qh = q + (size_t)bh * NN * DD;
  const u16* kh = k + (size_t)bh * NN * DD;
  const u16* vh = vt + (size_t)bh * DD * NN;

  // Q fragments for d-chunks [0,32) and [32,64)
  const u16* qrow = qh + (size_t)(qm0 + ln) * DD;
  v16bf aQ0 = load_a_frag(qrow + 0 + 16 * half);
  v16bf aQ1 = load_a_frag(qrow + 32 + 16 * half);

  float rm[8], rl[8];
  v8f oacc[4] = {};
#pragma unroll
  for (int r = 0; r < 8; ++r) { rm[r] = -1e30f; rl[r] = 0.f; }

#pragma unroll 1
  for (int kb0 = 0; kb0 < NN; kb0 += 32) {
    // ---- batch all K fragments, then the 4 QK^T WMMAs ----
    v16bf b00 = load_b_frag(kh + (size_t)(kb0 + ln) * DD + 0 + 16 * half);
    v16bf b01 = load_b_frag(kh + (size_t)(kb0 + 16 + ln) * DD + 0 + 16 * half);
    v16bf b10 = load_b_frag(kh + (size_t)(kb0 + ln) * DD + 32 + 16 * half);
    v16bf b11 = load_b_frag(kh + (size_t)(kb0 + 16 + ln) * DD + 32 + 16 * half);
    // prefetch V fragments now; not consumed until after the softmax VALU work
    v16bf bV0 = load_b_frag(vh + ((size_t)(0 * 16 + ln) << 10) + kb0 + 16 * half);
    v16bf bV1 = load_b_frag(vh + ((size_t)(1 * 16 + ln) << 10) + kb0 + 16 * half);
    v16bf bV2 = load_b_frag(vh + ((size_t)(2 * 16 + ln) << 10) + kb0 + 16 * half);
    v16bf bV3 = load_b_frag(vh + ((size_t)(3 * 16 + ln) << 10) + kb0 + 16 * half);
    float pol0 = policy[(b << 10) + kb0 + ln];
    float pol1 = policy[(b << 10) + kb0 + 16 + ln];

    v8f s0 = {}, s1 = {};
    s0 = wmma_bf16(aQ0, b00, s0);
    s1 = wmma_bf16(aQ0, b01, s1);
    s0 = wmma_bf16(aQ1, b10, s0);
    s1 = wmma_bf16(aQ1, b11, s1);

    // ---- online softmax (max over RAW scores, mask applied to exp) ----
    float p0[8], p1[8];
#pragma unroll
    for (int r = 0; r < 8; ++r) {
      float bm = rmax16(fmaxf(s0[r], s1[r]));
      float nm = fmaxf(rm[r], bm);
      float corr = __expf(rm[r] - nm);
      rm[r] = nm;
      rl[r] *= corr;
#pragma unroll
      for (int t = 0; t < 4; ++t) oacc[t][r] *= corr;
      int rowid = qm0 + 8 * half + r;
      float ap0 = (kb0 + ln == rowid) ? 1.0f : pol0;        // diag always allowed
      float ap1 = (kb0 + 16 + ln == rowid) ? 1.0f : pol1;
      p0[r] = __expf(s0[r] - nm) * ap0;
      p1[r] = __expf(s1[r] - nm) * ap1;
      rl[r] += rsum16(p0[r] + p1[r]);
    }

    // ---- C-layout f32 -> A-layout bf16 via LDS ----
#pragma unroll
    for (int r = 0; r < 8; ++r) {
      pt[(8 * half + r) * 32 + ln]      = f2bf(p0[r]);
      pt[(8 * half + r) * 32 + 16 + ln] = f2bf(p1[r]);
    }
    asm volatile("s_wait_dscnt 0" ::: "memory");   // wave-internal LDS RAW
    v16bf aP = load_a_frag(&pt[ln * 32 + 16 * half]);

    // ---- O += P V  (contraction over 32 keys per WMMA) ----
    oacc[0] = wmma_bf16(aP, bV0, oacc[0]);
    oacc[1] = wmma_bf16(aP, bV1, oacc[1]);
    oacc[2] = wmma_bf16(aP, bV2, oacc[2]);
    oacc[3] = wmma_bf16(aP, bV3, oacc[3]);
  }

  // ---- normalize and write [B,N,C] bf16 ----
#pragma unroll
  for (int r = 0; r < 8; ++r) {
    float inv = 1.0f / (rl[r] + EPS);
    int tok = qm0 + 8 * half + r;
#pragma unroll
    for (int t = 0; t < 4; ++t) {
      int c = h * DD + t * 16 + ln;
      ao[((size_t)(b << 10) + tok) * CC + c] = f2bf(oacc[t][r] * inv);
    }
  }
}

// ---------------- host launch ----------------
extern "C" void kernel_launch(void* const* d_in, const int* in_sizes, int n_in,
                              void* d_out, int out_size, void* d_ws, size_t ws_size,
                              hipStream_t stream) {
  const float* x      = (const float*)d_in[0];   // [8,1024,768]
  const float* policy = (const float*)d_in[1];   // [8,1024,1]
  const float* w_qkv  = (const float*)d_in[2];   // [2304,768]
  const float* w_proj = (const float*)d_in[3];   // [768,768]
  const float* b_proj = (const float*)d_in[4];   // [768]
  float* out = (float*)d_out;

  const size_t nX  = (size_t)BB * NN * CC;       // 6291456
  const size_t nWq = (size_t)3 * CC * CC;        // 1769472
  const size_t nWp = (size_t)CC * CC;            // 589824
  const size_t nQ  = (size_t)BB * HH * NN * DD;  // 6291456

  char* ws = (char*)d_ws;
  size_t off = 0;
  u16* xb     = (u16*)(ws + off); off += nX  * 2;
  u16* wqkvb  = (u16*)(ws + off); off += nWq * 2;
  u16* wprojb = (u16*)(ws + off); off += nWp * 2;
  u16* qbuf   = (u16*)(ws + off); off += nQ  * 2;
  u16* kbuf   = (u16*)(ws + off); off += nQ  * 2;
  u16* vtbuf  = (u16*)(ws + off); off += nQ  * 2;
  u16* aobuf  = (u16*)(ws + off); off += nX  * 2;   // total ~67.6 MB

  cast_bf16_kernel<<<(int)((nX  + 255) / 256), 256, 0, stream>>>(x, xb, (int)nX);
  cast_bf16_kernel<<<(int)((nWq + 255) / 256), 256, 0, stream>>>(w_qkv, wqkvb, (int)nWq);
  cast_bf16_kernel<<<(int)((nWp + 255) / 256), 256, 0, stream>>>(w_proj, wprojb, (int)nWp);

  gemm_qkv_kernel<<<dim3(128, 36), 128, 0, stream>>>(xb, wqkvb, qbuf, kbuf, vtbuf);
  attn_kernel<<<dim3(64, 96), 32, 0, stream>>>(qbuf, kbuf, vtbuf, policy, aobuf);
  gemm_proj_kernel<<<dim3(128, 12), 128, 0, stream>>>(aobuf, wprojb, b_proj, out);
}
